// Attention_25744033973106
// MI455X (gfx1250) — matrix operand order repaired and checked by reading, not compiled, not run on hardware
//
#include <hip/hip_runtime.h>

typedef __attribute__((ext_vector_type(16))) __bf16 v16bf;
typedef __attribute__((ext_vector_type(8)))  float  v8f;
typedef __attribute__((ext_vector_type(4)))  int    v4i;

#define GLOBAL_AS __attribute__((address_space(1)))
#define SHARED_AS __attribute__((address_space(3)))

#define DIM    512
#define HEADS  8
#define HD     64
#define NTOK   4096
#define BATCH  2
#define MROWS  (BATCH * NTOK)   // 8192

#if defined(__has_builtin)
# if __has_builtin(__builtin_amdgcn_global_load_async_to_lds_b128)
#  define HAVE_ASYNC_LDS 1
# endif
# if __has_builtin(__builtin_amdgcn_s_wait_asynccnt)
#  define HAVE_WAIT_ASYNC 1
# endif
#endif

__device__ __forceinline__ void wait_async_n0() {
#if defined(HAVE_WAIT_ASYNC)
  __builtin_amdgcn_s_wait_asynccnt(0);
#elif defined(HAVE_ASYNC_LDS)
  asm volatile("s_wait_asynccnt 0x0" ::: "memory");
#endif
}
__device__ __forceinline__ void wait_async_n1() {
#if defined(HAVE_WAIT_ASYNC)
  __builtin_amdgcn_s_wait_asynccnt(1);
#elif defined(HAVE_ASYNC_LDS)
  asm volatile("s_wait_asynccnt 0x1" ::: "memory");
#endif
}
__device__ __forceinline__ void wait_async_n2() {
#if defined(HAVE_WAIT_ASYNC)
  __builtin_amdgcn_s_wait_asynccnt(2);
#elif defined(HAVE_ASYNC_LDS)
  asm volatile("s_wait_asynccnt 0x2" ::: "memory");
#endif
}

#if defined(HAVE_ASYNC_LDS)
__device__ __forceinline__ void async_copy16(const void* g, void* l) {
  const v4i* gp = (const v4i*)g;   // generic pointer, v4i pointee
  v4i* lp = (v4i*)l;
  __builtin_amdgcn_global_load_async_to_lds_b128(
      (GLOBAL_AS v4i*)gp,          // const drop + addrspace(1)
      (SHARED_AS v4i*)lp,          // addrspace(3)
      0, 0);
}
#endif

// Fragment viewed as two 16-byte halves for vector LDS/global access.
union alignas(32) BFrag {
  v16bf v;
  uint4 q[2];
};

// ---------------------------------------------------------------------------
// fp32 -> bf16 conversion (vector-4), row-major preserved (for x)
// ---------------------------------------------------------------------------
__global__ void cvt_f32_bf16_v4(const float* __restrict__ src,
                                __bf16* __restrict__ dst, int n4) {
  int i = blockIdx.x * blockDim.x + threadIdx.x;
  if (i < n4) {
    float4 f = ((const float4*)src)[i];
    __bf16* d = dst + 4 * i;
    d[0] = (__bf16)f.x;
    d[1] = (__bf16)f.y;
    d[2] = (__bf16)f.z;
    d[3] = (__bf16)f.w;
  }
}

// ---------------------------------------------------------------------------
// fp32 [K][N] -> bf16 [N][K] transpose (for weights). Block 256 = 32x8,
// 32x32 tile via LDS. Grid (16,16).
// ---------------------------------------------------------------------------
__global__ __launch_bounds__(256)
void cvt_w_transpose(const float* __restrict__ src, __bf16* __restrict__ dst) {
  __shared__ float tile[32][33];
  const int bx = blockIdx.x * 32;   // N block
  const int by = blockIdx.y * 32;   // K block
  const int tx = threadIdx.x & 31;
  const int ty = threadIdx.x >> 5;  // 0..7
#pragma unroll
  for (int j = 0; j < 32; j += 8)
    tile[ty + j][tx] = src[(size_t)(by + ty + j) * DIM + bx + tx];
  __syncthreads();
#pragma unroll
  for (int j = 0; j < 32; j += 8)
    dst[(size_t)(bx + ty + j) * DIM + by + tx] = (__bf16)tile[tx][ty + j];
}

// ---------------------------------------------------------------------------
// GEMM: C[M=8192,N=512] = A[8192,512] * B + bias, with B given TRANSPOSED
// as BT[N=512][K=512] bf16 (row-major, K contiguous).
// mode 0: bf16 scatter into [BATCH, HEADS, NTOK, HD]        (Q, K)
// mode 2: bf16 scatter into [BATCH, HEADS, HD, NTOK]        (V, transposed)
// mode 1: fp32 row-major [8192,512]                         (final output)
// Block: 256 threads = 8 waves; block tile 128(M) x 64(N); wave tile 16x64.
// B tiles double-buffered through LDS via async-to-LDS.
// ---------------------------------------------------------------------------
__global__ __launch_bounds__(256)
void gemm_bf16(const __bf16* __restrict__ A, const __bf16* __restrict__ BT,
               const float* __restrict__ bias, __bf16* __restrict__ outb,
               float* __restrict__ outf, int mode) {
  __shared__ __bf16 BsT[2][64 * 32];   // [n][k] tiles, k contiguous

  const int tid  = threadIdx.x;
  const int wave = tid >> 5;
  const int lane = tid & 31;
  const int half = lane >> 4;
  const int ln   = lane & 15;
  const int mbase = blockIdx.x * 128 + wave * 16;
  const int nbase = blockIdx.y * 64;

  const int br = tid >> 2;          // 0..63 (n)
  const int bc = (tid & 3) * 8;     // 0..24 (k)

  auto stageB = [&](int k0, int buf) {
#if defined(HAVE_ASYNC_LDS)
    async_copy16(BT + (size_t)(nbase + br) * DIM + k0 + bc, &BsT[buf][br * 32 + bc]);
#else
    *(uint4*)&BsT[buf][br * 32 + bc] =
        *(const uint4*)(BT + (size_t)(nbase + br) * DIM + k0 + bc);
#endif
  };

  v8f acc[4] = {};

  stageB(0, 0);
  for (int k0 = 0; k0 < DIM; k0 += 32) {
    const int buf = (k0 >> 5) & 1;
    const bool more = (k0 + 32) < DIM;
    if (more) stageB(k0 + 32, buf ^ 1);   // prefetch next tile
#if defined(HAVE_ASYNC_LDS)
    if (more) wait_async_n1(); else wait_async_n0();   // oldest tile landed
#endif
    __syncthreads();

    // A fragment (16x32 k-slice), two global 16B loads
    BFrag a;
    {
      const __bf16* arow = A + (size_t)(mbase + ln) * DIM + k0 + half * 8;
      a.q[0] = *(const uint4*)(arow);
      a.q[1] = *(const uint4*)(arow + 16);
    }

#pragma unroll
    for (int nb = 0; nb < 4; ++nb) {
      BFrag b;   // lane ln holds column nb*16+ln, K contiguous
      const __bf16* bp = &BsT[buf][(nb * 16 + ln) * 32 + half * 8];
      b.q[0] = *(const uint4*)(bp);
      b.q[1] = *(const uint4*)(bp + 16);
      acc[nb] = __builtin_amdgcn_wmma_f32_16x16x32_bf16(
          false, a.v, false, b.v, (short)0, acc[nb], false, false);
    }
    __syncthreads();   // all waves done with buf before it is restaged
  }

  // epilogue: bias + scatter
#pragma unroll
  for (int nb = 0; nb < 4; ++nb) {
    const int col = nbase + nb * 16 + ln;
    const float bv = bias[col];
#pragma unroll
    for (int j = 0; j < 8; ++j) {
      const int row = mbase + half * 8 + j;
      const float v = acc[nb][j] + bv;
      const int batch = row >> 12;     // /4096
      const int tok   = row & 4095;
      const int head  = col >> 6;      // /64
      const int hdix  = col & 63;
      if (mode == 0) {
        outb[(((size_t)(batch * HEADS + head)) * NTOK + tok) * HD + hdix] = (__bf16)v;
      } else if (mode == 2) {
        outb[(((size_t)(batch * HEADS + head)) * HD + hdix) * NTOK + tok] = (__bf16)v;
      } else {
        outf[(size_t)row * DIM + col] = v;
      }
    }
  }
}

// ---------------------------------------------------------------------------
// Flash attention with Euclidean-distance ALiBi bias.
// Q,K in [b,h,n,64]; V in [b,h,64,n] (transposed). Grid (NTOK/128, 16).
// Block 256 = 8 waves; each wave owns 16 query rows; 32-key LDS tiles,
// double-buffered via async-to-LDS.
// ---------------------------------------------------------------------------
__global__ __launch_bounds__(256)
void flash_attn(const __bf16* __restrict__ Q, const __bf16* __restrict__ K,
                const __bf16* __restrict__ VT, const float* __restrict__ coords,
                const float* __restrict__ slopes, __bf16* __restrict__ O) {
  __shared__ __bf16 Ks[2][32 * 64];    // [key][hd]
  __shared__ __bf16 VsT[2][64 * 32];   // [hd][key]
  __shared__ float  Cs[2][64];
  __shared__ __bf16 Ps[8 * 16 * 32];   // per-wave P staging (C->A relayout)

  const int tid  = threadIdx.x;
  const int wave = tid >> 5;
  const int lane = tid & 31;
  const int half = lane >> 4;
  const int ln   = lane & 15;

  const int bh    = blockIdx.y;        // 0..15
  const int batch = bh >> 3;
  const int head  = bh & (HEADS - 1);
  const int q0    = blockIdx.x * 128 + wave * 16;

  const float slope = slopes[head];
  const float scale = 0.125f;          // 1/sqrt(64)

  const __bf16* Qh = Q + (size_t)bh * NTOK * HD;
  const __bf16* Kh = K + (size_t)bh * NTOK * HD;
  const __bf16* Vh = VT + (size_t)bh * HD * NTOK;
  const float*  cb = coords + (size_t)batch * NTOK * 2;
  __bf16* pw = Ps + wave * (16 * 32);

  // Q fragments (k-slices [0,32) and [32,64)) via 16B global loads
  BFrag qa0, qa1;
  {
    const __bf16* qrow = Qh + (size_t)(q0 + ln) * HD + half * 8;
    qa0.q[0] = *(const uint4*)(qrow);
    qa0.q[1] = *(const uint4*)(qrow + 16);
    qa1.q[0] = *(const uint4*)(qrow + 32);
    qa1.q[1] = *(const uint4*)(qrow + 48);
  }

  // per-lane query coords for the 8 C-fragment rows (M = half*8 + j)
  float qcx[8], qcy[8];
#pragma unroll
  for (int j = 0; j < 8; ++j) {
    const int r = q0 + half * 8 + j;
    qcx[j] = cb[2 * r];
    qcy[j] = cb[2 * r + 1];
  }

  v8f o[4] = {};
  float m[8], l[8];
#pragma unroll
  for (int j = 0; j < 8; ++j) { m[j] = -1e30f; l[j] = 0.0f; }

  // staging addresses for this thread
  const int kr = tid >> 3;            // 0..31 (key row for Ks)
  const int kc = (tid & 7) * 8;       // 0..56 (hd col for Ks)
  const int vr = tid >> 2;            // 0..63 (hd row for VsT)
  const int vc = (tid & 3) * 8;       // 0..24 (key col for VsT)

  auto stage = [&](int kb, int buf) {
#if defined(HAVE_ASYNC_LDS)
    async_copy16(Kh + (size_t)(kb + kr) * HD + kc, &Ks[buf][kr * 64 + kc]);
    async_copy16(Vh + (size_t)vr * NTOK + kb + vc, &VsT[buf][vr * 32 + vc]);
#else
    *(uint4*)&Ks[buf][kr * 64 + kc]  = *(const uint4*)(Kh + (size_t)(kb + kr) * HD + kc);
    *(uint4*)&VsT[buf][vr * 32 + vc] = *(const uint4*)(Vh + (size_t)vr * NTOK + kb + vc);
#endif
    if (tid < 64) Cs[buf][tid] = cb[(size_t)kb * 2 + tid];
  };

  stage(0, 0);
  for (int kb = 0; kb < NTOK; kb += 32) {
    const int buf  = (kb >> 5) & 1;
    const bool more = (kb + 32) < NTOK;
    if (more) stage(kb + 32, buf ^ 1);   // prefetch next K/V tile
#if defined(HAVE_ASYNC_LDS)
    // this wave issued 2 async loads per tile; async loads complete in order,
    // so asynccnt<=2 means the current tile's loads have landed.
    if (more) wait_async_n2(); else wait_async_n0();
#endif
    __syncthreads();

    // ---- S = Q * K^T : two 16x16 tiles over this 32-key block ----
    v8f s[2] = {};
#pragma unroll
    for (int t = 0; t < 2; ++t) {
      // B frag: lane ln = key t*16+ln, K (hd) contiguous
      const __bf16* kp = &Ks[buf][(t * 16 + ln) * 64 + half * 8];
      BFrag b0, b1;
      b0.q[0] = *(const uint4*)(kp);
      b0.q[1] = *(const uint4*)(kp + 16);
      b1.q[0] = *(const uint4*)(kp + 32);
      b1.q[1] = *(const uint4*)(kp + 48);
      s[t] = __builtin_amdgcn_wmma_f32_16x16x32_bf16(
          false, qa0.v, false, b0.v, (short)0, s[t], false, false);
      s[t] = __builtin_amdgcn_wmma_f32_16x16x32_bf16(
          false, qa1.v, false, b1.v, (short)0, s[t], false, false);
    }

    // ---- scale + distance-ALiBi bias ----
#pragma unroll
    for (int t = 0; t < 2; ++t) {
      const float kx = Cs[buf][(t * 16 + ln) * 2];
      const float ky = Cs[buf][(t * 16 + ln) * 2 + 1];
#pragma unroll
      for (int j = 0; j < 8; ++j) {
        const float dx = qcx[j] - kx;
        const float dy = qcy[j] - ky;
        s[t][j] = s[t][j] * scale - slope * __fsqrt_rn(dx * dx + dy * dy);
      }
    }

    // ---- online softmax (rows live in 16-lane halves) ----
#pragma unroll
    for (int j = 0; j < 8; ++j) {
      float t = fmaxf(s[0][j], s[1][j]);
#pragma unroll
      for (int off = 1; off < 16; off <<= 1)
        t = fmaxf(t, __shfl_xor(t, off, 16));
      const float mn   = fmaxf(m[j], t);
      const float corr = __expf(m[j] - mn);
      m[j] = mn;
      const float p0 = __expf(s[0][j] - mn);
      const float p1 = __expf(s[1][j] - mn);
      s[0][j] = p0;
      s[1][j] = p1;
      float rs = p0 + p1;
#pragma unroll
      for (int off = 1; off < 16; off <<= 1)
        rs += __shfl_xor(rs, off, 16);
      l[j] = l[j] * corr + rs;
#pragma unroll
      for (int nb = 0; nb < 4; ++nb) o[nb][j] *= corr;
    }

    // ---- P: C-fragment -> A-fragment relayout through per-wave LDS ----
#pragma unroll
    for (int j = 0; j < 8; ++j) {
      pw[(half * 8 + j) * 32 + ln]      = (__bf16)s[0][j];
      pw[(half * 8 + j) * 32 + 16 + ln] = (__bf16)s[1][j];
    }
    // same-wave DS ops are in-order; no cross-wave sharing -> no barrier
    BFrag pa;
    {
      const __bf16* pp = &pw[ln * 32 + half * 8];
      pa.q[0] = *(const uint4*)(pp);
      pa.q[1] = *(const uint4*)(pp + 16);
    }

    // ---- O += P * V ----
#pragma unroll
    for (int nb = 0; nb < 4; ++nb) {
      BFrag vb;   // lane ln = hd col nb*16+ln, key contiguous
      const __bf16* vp = &VsT[buf][(nb * 16 + ln) * 32 + half * 8];
      vb.q[0] = *(const uint4*)(vp);
      vb.q[1] = *(const uint4*)(vp + 16);
      o[nb] = __builtin_amdgcn_wmma_f32_16x16x32_bf16(
          false, pa.v, false, vb.v, (short)0, o[nb], false, false);
    }

    __syncthreads();   // all waves done with buf before it is restaged
  }

  // ---- epilogue: normalize and write O as bf16 [b, n, d] ----
#pragma unroll
  for (int j = 0; j < 8; ++j) {
    const float inv = 1.0f / l[j];
    const int row = q0 + half * 8 + j;
#pragma unroll
    for (int nb = 0; nb < 4; ++nb) {
      const int col = head * HD + nb * 16 + ln;
      O[((size_t)batch * NTOK + row) * DIM + col] = (__bf16)(o[nb][j] * inv);
    }
  }
}

// ---------------------------------------------------------------------------
// Launch
// ---------------------------------------------------------------------------
extern "C" void kernel_launch(void* const* d_in, const int* in_sizes, int n_in,
                              void* d_out, int out_size, void* d_ws, size_t ws_size,
                              hipStream_t stream) {
  const float* x      = (const float*)d_in[0];
  const float* coords = (const float*)d_in[1];
  const float* Wq     = (const float*)d_in[2];
  const float* bq     = (const float*)d_in[3];
  const float* Wk     = (const float*)d_in[4];
  const float* bk     = (const float*)d_in[5];
  const float* Wv     = (const float*)d_in[6];
  const float* bv     = (const float*)d_in[7];
  const float* Wp     = (const float*)d_in[8];
  const float* bp     = (const float*)d_in[9];
  const float* slopes = (const float*)d_in[10];
  float* out = (float*)d_out;

  char* ws = (char*)d_ws;
  size_t off = 0;
  auto alloc = [&](size_t bytes) -> void* {
    void* p = ws + off;
    off += (bytes + 255) & ~(size_t)255;
    return p;
  };

  const size_t xelems = (size_t)MROWS * DIM;   // 4,194,304
  const size_t welems = (size_t)DIM * DIM;     //   262,144

  __bf16* Xb   = (__bf16*)alloc(xelems * 2);
  __bf16* WqT  = (__bf16*)alloc(welems * 2);
  __bf16* WkT  = (__bf16*)alloc(welems * 2);
  __bf16* WvT  = (__bf16*)alloc(welems * 2);
  __bf16* WpT  = (__bf16*)alloc(welems * 2);
  __bf16* Qb   = (__bf16*)alloc(xelems * 2);
  __bf16* Kb   = (__bf16*)alloc(xelems * 2);
  __bf16* Vb   = (__bf16*)alloc(xelems * 2);
  __bf16* Ob   = (__bf16*)alloc(xelems * 2);

  // x: straight convert; W: convert + transpose to [N][K]
  {
    int n4 = (int)(xelems / 4);
    cvt_f32_bf16_v4<<<(n4 + 255) / 256, 256, 0, stream>>>(x, Xb, n4);
    dim3 gt(DIM / 32, DIM / 32);
    cvt_w_transpose<<<gt, 256, 0, stream>>>(Wq, WqT);
    cvt_w_transpose<<<gt, 256, 0, stream>>>(Wk, WkT);
    cvt_w_transpose<<<gt, 256, 0, stream>>>(Wv, WvT);
    cvt_w_transpose<<<gt, 256, 0, stream>>>(Wp, WpT);
  }

  // Q/K/V projections (V stored transposed per-head)
  dim3 gg(MROWS / 128, DIM / 64);
  gemm_bf16<<<gg, 256, 0, stream>>>(Xb, WqT, bq, Qb, nullptr, 0);
  gemm_bf16<<<gg, 256, 0, stream>>>(Xb, WkT, bk, Kb, nullptr, 0);
  gemm_bf16<<<gg, 256, 0, stream>>>(Xb, WvT, bv, Vb, nullptr, 2);

  // flash attention
  dim3 ga(NTOK / 128, BATCH * HEADS);
  flash_attn<<<ga, 256, 0, stream>>>(Qb, Kb, Vb, coords, slopes, Ob);

  // output projection -> fp32
  gemm_bf16<<<gg, 256, 0, stream>>>(Ob, WpT, bp, nullptr, out, 1);
}
